// Segmenter_22179211117076
// MI455X (gfx1250) — matrix operand order repaired and checked by reading, not compiled, not run on hardware
//
#include <hip/hip_runtime.h>
#include <hip/hip_bf16.h>
#include <math.h>

// Problem constants (from reference)
#define VV 512
#define EE 128
#define HH 256
#define LL 4
#define BB 128
#define TT 1024

typedef __attribute__((ext_vector_type(16))) __bf16 v16bf;
typedef __attribute__((ext_vector_type(8)))  __bf16 v8bf;
typedef __attribute__((ext_vector_type(8)))  float  v8f;

__device__ __forceinline__ v8f vsplat8f(float x) {
    v8f v;
#pragma unroll
    for (int e = 0; e < 8; ++e) v[e] = x;
    return v;
}

__device__ __forceinline__ v8f wmma_bf16(v16bf a, v16bf b, v8f c) {
    // D = A(16x32 bf16) * B(32x16 bf16) + C(16x16 f32)
    return __builtin_amdgcn_wmma_f32_16x16x32_bf16(
        false, a, false, b, (short)0, c, false, false);
}

__device__ __forceinline__ float sigmoidf_(float x) {
    return 1.0f / (1.0f + expf(-x));
}

// A-fragment (16x32 bf16) from a bf16 row in global memory (xe row).
// Lane (m = lane&15, kh = lane>>4): elems 0..7 -> k = kb*32 + kh*8 + e,
// elems 8..15 -> k = kb*32 + 16 + kh*8 + e. Two contiguous 16B chunks.
__device__ __forceinline__ v16bf load_afrag_xe(const __bf16* __restrict__ xrow,
                                               int kb, int kh) {
    int k0 = kb * 32 + kh * 8;
    v8bf lo = *(const v8bf*)(xrow + k0);
    v8bf hi = *(const v8bf*)(xrow + k0 + 16);
    return __builtin_shufflevector(lo, hi, 0, 1, 2, 3, 4, 5, 6, 7,
                                   8, 9, 10, 11, 12, 13, 14, 15);
}

// A-fragment from the f32 hidden state row in LDS (convert to bf16).
__device__ __forceinline__ v16bf load_afrag_h(const float* hrow, int kb, int kh) {
    int k0 = kb * 32 + kh * 8;
    v16bf f;
#pragma unroll
    for (int e = 0; e < 8; ++e) {
        f[e]     = (__bf16)hrow[k0 + e];
        f[e + 8] = (__bf16)hrow[k0 + 16 + e];
    }
    return f;
}

// ---------------------------------------------------------------------------
// Kernel 0: pre-swizzle a (3H x rowStride) f32 weight matrix into bf16 WMMA
// B-fragment blobs. Fragment f = (g*16 + wv)*KB + kb; per lane 16 contiguous
// bf16 (32B). One thread per (fragment, lane).
// ---------------------------------------------------------------------------
__global__ void swizzle_w_kernel(const float* __restrict__ w,
                                 __bf16* __restrict__ frag,
                                 int KB, int rowStride) {
    int idx = blockIdx.x * blockDim.x + threadIdx.x;
    int total = 3 * 16 * KB * 32;
    if (idx >= total) return;
    int lane = idx & 31;
    int f    = idx >> 5;
    int kb   = f % KB;
    int gw   = f / KB;
    int wv   = gw & 15;
    int g    = gw >> 4;
    int grow = g * HH + wv * 16 + (lane & 15);
    int k0   = kb * 32 + (lane >> 4) * 8;
    const float* p = w + (size_t)grow * rowStride + k0;
    __bf16* d = frag + ((size_t)f * 32 + lane) * 16;
#pragma unroll
    for (int e = 0; e < 16; ++e) d[e] = (__bf16)p[e < 8 ? e : e + 8];
}

// ---------------------------------------------------------------------------
// Kernel 1: gather embeddings to bf16  xe[b,t,:] = bf16(emb[x[b,t],:])
// ---------------------------------------------------------------------------
__global__ void embed_kernel(const int* __restrict__ x,
                             const float* __restrict__ emb,
                             __bf16* __restrict__ xe) {
    int i = blockIdx.x * blockDim.x + threadIdx.x;
    if (i >= BB * TT) return;
    int v = x[i];
    const float* src = emb + (size_t)v * EE;
    __bf16* dst = xe + (size_t)i * EE;
#pragma unroll 8
    for (int e = 0; e < EE; ++e) dst[e] = (__bf16)src[e];
}

// ---------------------------------------------------------------------------
// Kernel 2: zero the emissions accumulator
// ---------------------------------------------------------------------------
__global__ void zero_kernel(float* __restrict__ p, int n) {
    int i = blockIdx.x * blockDim.x + threadIdx.x;
    if (i < n) p[i] = 0.0f;
}

// ---------------------------------------------------------------------------
// Kernel 3: persistent BiGRU recurrence with fused emission projection.
// 16 WGs (dir x 8 batch tiles) x 512 threads (16 waves). Wave w owns gate
// columns {16w..16w+15} of r,z,n. Weight B-fragments are STREAMED from the
// pre-swizzled bf16 blobs (L2-resident, shared across the 8 WGs per dir).
// The anti-LICM asm barrier acts on a zero OFFSET (not the pointers) so the
// loads stay in the global address space (global_load_b128, LOADcnt only)
// instead of degrading to flat_load (LOADcnt+DScnt, LDS-pipe coupled).
// ---------------------------------------------------------------------------
__global__ __launch_bounds__(512)
void gru_kernel(const __bf16* __restrict__ xe,
                const __bf16* __restrict__ wihFragF, const __bf16* __restrict__ whhFragF,
                const __bf16* __restrict__ wihFragB, const __bf16* __restrict__ whhFragB,
                const float* __restrict__ bih_f, const float* __restrict__ bhh_f,
                const float* __restrict__ bih_b, const float* __restrict__ bhh_b,
                const float* __restrict__ fcw,   // (L, 2H) row-major
                float* __restrict__ em)          // (B, T, L) accumulator
{
    __shared__ float hbuf[16][HH + 16];   // hidden state, f32, padded rows
    __shared__ float fcwl[LL][HH];        // fc_w slice for this direction

    const int tid  = threadIdx.x;
    const int dir  = blockIdx.x >> 3;
    const int rb   = blockIdx.x & 7;
    const int lane = tid & 31;
    const int wv   = tid >> 5;            // wave id 0..15
    const int nIdx = lane & 15;
    const int kh   = lane >> 4;

    const __bf16* wihFrag = dir ? wihFragB : wihFragF;
    const __bf16* whhFrag = dir ? whhFragB : whhFragF;
    const float*  bih     = dir ? bih_b : bih_f;
    const float*  bhh     = dir ? bhh_b : bhh_f;

    // init: h = 0, stage fc_w slice
    for (int i = tid; i < 16 * (HH + 16); i += 512) (&hbuf[0][0])[i] = 0.0f;
    for (int i = tid; i < LL * HH; i += 512) {
        int l = i / HH, k = i - l * HH;
        fcwl[l][k] = fcw[l * (2 * HH) + dir * HH + k];
    }

    // bias seeds for accumulators (col = wv*16 + nIdx per gate)
    const float bR  = bih[0 * HH + wv * 16 + nIdx] + bhh[0 * HH + wv * 16 + nIdx];
    const float bZ  = bih[1 * HH + wv * 16 + nIdx] + bhh[1 * HH + wv * 16 + nIdx];
    const float bNi = bih[2 * HH + wv * 16 + nIdx];
    const float bNh = bhh[2 * HH + wv * 16 + nIdx];

    // per-wave/lane fragment base pointers into the swizzled blobs.
    // elem offset of frag (g,kb): (g*16*KB + wv*KB + kb)*512 + lane*16
    const __bf16* __restrict__ wihBase0 = wihFrag + ((size_t)wv * 4) * 512 + (size_t)lane * 16;
    const __bf16* __restrict__ whhBase0 = whhFrag + ((size_t)wv * 8) * 512 + (size_t)lane * 16;

    __syncthreads();

    const int bRow = rb * 16 + nIdx;      // batch row this lane's A-frags use
    const int b0   = rb * 16;
    const int col  = wv * 16 + nIdx;      // C-layout column for this lane

    for (int s = 0; s < TT; ++s) {
        const int t = dir ? (TT - 1 - s) : s;

        // Opaque zero offset: defeats LICM on the (loop-invariant) weight
        // fragment addresses without erasing the global address space.
        unsigned long long zoff = 0;
        asm volatile("" : "+v"(zoff));
        const __bf16* __restrict__ wihB = wihBase0 + zoff;
        const __bf16* __restrict__ whhB = whhBase0 + zoff;

        // A-fragments: xe tile (global bf16) and h tile (LDS f32 -> bf16)
        const __bf16* xrow = xe + ((size_t)bRow * TT + t) * EE;
        v16bf Ax[4];
#pragma unroll
        for (int kb = 0; kb < 4; ++kb) Ax[kb] = load_afrag_xe(xrow, kb, kh);
        v16bf Ah[8];
#pragma unroll
        for (int kb = 0; kb < 8; ++kb) Ah[kb] = load_afrag_h(&hbuf[nIdx][0], kb, kh);

        // accumulators seeded with biases; r/z chain input+hidden GEMMs
        v8f accR  = vsplat8f(bR);
        v8f accZ  = vsplat8f(bZ);
        v8f accNi = vsplat8f(bNi);
        v8f accNh = vsplat8f(bNh);

#pragma unroll
        for (int kb = 0; kb < 4; ++kb)
            accR  = wmma_bf16(Ax[kb], *(const v16bf*)(wihB + (0 * 64 + kb) * 512), accR);
#pragma unroll
        for (int kb = 0; kb < 8; ++kb)
            accR  = wmma_bf16(Ah[kb], *(const v16bf*)(whhB + (0 * 128 + kb) * 512), accR);
#pragma unroll
        for (int kb = 0; kb < 4; ++kb)
            accZ  = wmma_bf16(Ax[kb], *(const v16bf*)(wihB + (1 * 64 + kb) * 512), accZ);
#pragma unroll
        for (int kb = 0; kb < 8; ++kb)
            accZ  = wmma_bf16(Ah[kb], *(const v16bf*)(whhB + (1 * 128 + kb) * 512), accZ);
#pragma unroll
        for (int kb = 0; kb < 4; ++kb)
            accNi = wmma_bf16(Ax[kb], *(const v16bf*)(wihB + (2 * 64 + kb) * 512), accNi);
#pragma unroll
        for (int kb = 0; kb < 8; ++kb)
            accNh = wmma_bf16(Ah[kb], *(const v16bf*)(whhB + (2 * 128 + kb) * 512), accNh);

        // GRU cell (C-layout: VGPR v -> row v + kh*8, col = wv*16 + lane&15)
        float hnew[8];
#pragma unroll
        for (int v = 0; v < 8; ++v) {
            int m = v + kh * 8;
            float r  = sigmoidf_(accR[v]);
            float z  = sigmoidf_(accZ[v]);
            float nn = tanhf(accNi[v] + r * accNh[v]);
            float ho = hbuf[m][col];
            hnew[v]  = (1.0f - z) * nn + z * ho;
        }

        __syncthreads();   // all reads of h done -> safe to overwrite
#pragma unroll
        for (int v = 0; v < 8; ++v) hbuf[v + kh * 8][col] = hnew[v];
        __syncthreads();   // h updated -> visible to emission + next step

        // fused emission: em[b0+row, t, l] += h . fcw[l, dirH:], split-K x2
        if (tid < 128) {
            int row = tid >> 3, l = (tid >> 1) & 3, k2 = tid & 1;
            float sum = 0.0f;
#pragma unroll 8
            for (int k = k2 * 128; k < k2 * 128 + 128; ++k)
                sum += hbuf[row][k] * fcwl[l][k];
            atomicAdd(&em[(((size_t)(b0 + row)) * TT + t) * LL + l], sum);
        }
    }
}

// ---------------------------------------------------------------------------
// Kernel 4: CRF negative log-likelihood (mask all-true per setup).
// ---------------------------------------------------------------------------
__global__ __launch_bounds__(128)
void crf_kernel(const float* __restrict__ em,
                const int* __restrict__ tags,
                const float* __restrict__ fcb,
                const float* __restrict__ start_trans,
                const float* __restrict__ end_trans,
                const float* __restrict__ trans,
                float* __restrict__ out)
{
    __shared__ float red[BB];
    const int b = threadIdx.x;

    float st[LL], et[LL], fb[LL], tr[LL * LL];
#pragma unroll
    for (int l = 0; l < LL; ++l) { st[l] = start_trans[l]; et[l] = end_trans[l]; fb[l] = fcb[l]; }
#pragma unroll
    for (int i = 0; i < LL * LL; ++i) tr[i] = trans[i];

    const float* Eb = em + (size_t)b * TT * LL;
    const int*   tg = tags + (size_t)b * TT;

    int prev = tg[0];
    float score = st[prev] + Eb[prev] + fb[prev];
    float alpha[LL];
#pragma unroll
    for (int l = 0; l < LL; ++l) alpha[l] = st[l] + Eb[l] + fb[l];

    for (int t = 1; t < TT; ++t) {
        int cur = tg[t];
        const float* Et = Eb + (size_t)t * LL;
        score += tr[prev * LL + cur] + Et[cur] + fb[cur];
        prev = cur;

        float na[LL];
#pragma unroll
        for (int j = 0; j < LL; ++j) {
            float mx = -1e30f;
#pragma unroll
            for (int i = 0; i < LL; ++i) mx = fmaxf(mx, alpha[i] + tr[i * LL + j]);
            float ss = 0.0f;
#pragma unroll
            for (int i = 0; i < LL; ++i) ss += expf(alpha[i] + tr[i * LL + j] - mx);
            na[j] = mx + logf(ss) + Et[j] + fb[j];
        }
#pragma unroll
        for (int j = 0; j < LL; ++j) alpha[j] = na[j];
    }

    score += et[prev];
    float mx = -1e30f;
#pragma unroll
    for (int l = 0; l < LL; ++l) mx = fmaxf(mx, alpha[l] + et[l]);
    float ss = 0.0f;
#pragma unroll
    for (int l = 0; l < LL; ++l) ss += expf(alpha[l] + et[l] - mx);
    float logZ = mx + logf(ss);

    red[b] = score - logZ;
    __syncthreads();
    for (int off = BB / 2; off > 0; off >>= 1) {
        if (b < off) red[b] += red[b + off];
        __syncthreads();
    }
    if (b == 0) out[0] = -red[0] / (float)BB;
}

// ---------------------------------------------------------------------------
extern "C" void kernel_launch(void* const* d_in, const int* in_sizes, int n_in,
                              void* d_out, int out_size, void* d_ws, size_t ws_size,
                              hipStream_t stream) {
    const int*   x      = (const int*)  d_in[0];
    const int*   tags   = (const int*)  d_in[1];
    // d_in[2] = mask (all true per setup) — not needed
    const float* emb    = (const float*)d_in[3];
    const float* wih_f  = (const float*)d_in[4];
    const float* whh_f  = (const float*)d_in[5];
    const float* bih_f  = (const float*)d_in[6];
    const float* bhh_f  = (const float*)d_in[7];
    const float* wih_b  = (const float*)d_in[8];
    const float* whh_b  = (const float*)d_in[9];
    const float* bih_b  = (const float*)d_in[10];
    const float* bhh_b  = (const float*)d_in[11];
    const float* fcw    = (const float*)d_in[12];
    const float* fcb    = (const float*)d_in[13];
    const float* strt   = (const float*)d_in[14];
    const float* endt   = (const float*)d_in[15];
    const float* trans  = (const float*)d_in[16];

    // workspace layout
    char* ws = (char*)d_ws;
    const size_t XE_BYTES  = (size_t)BB * TT * EE * sizeof(__bf16);  // 33.5 MB
    const size_t EM_BYTES  = (size_t)BB * TT * LL * sizeof(float);   //  2.1 MB
    const size_t WIH_BYTES = (size_t)3 * 16 * 4 * 32 * 16 * sizeof(__bf16); // 192 KB
    const size_t WHH_BYTES = (size_t)3 * 16 * 8 * 32 * 16 * sizeof(__bf16); // 384 KB

    __bf16* xe    = (__bf16*)ws;                         ws += XE_BYTES;
    float*  em    = (float*)ws;                          ws += EM_BYTES;
    __bf16* wihFF = (__bf16*)ws;                         ws += WIH_BYTES;
    __bf16* wihFB = (__bf16*)ws;                         ws += WIH_BYTES;
    __bf16* whhFF = (__bf16*)ws;                         ws += WHH_BYTES;
    __bf16* whhFB = (__bf16*)ws;                         ws += WHH_BYTES;

    // 0) pre-swizzle weights into bf16 WMMA B-fragment blobs
    {
        const int nIH = 3 * 16 * 4 * 32;   // 6144 threads
        const int nHH = 3 * 16 * 8 * 32;   // 12288 threads
        swizzle_w_kernel<<<(nIH + 255) / 256, 256, 0, stream>>>(wih_f, wihFF, 4, EE);
        swizzle_w_kernel<<<(nIH + 255) / 256, 256, 0, stream>>>(wih_b, wihFB, 4, EE);
        swizzle_w_kernel<<<(nHH + 255) / 256, 256, 0, stream>>>(whh_f, whhFF, 8, HH);
        swizzle_w_kernel<<<(nHH + 255) / 256, 256, 0, stream>>>(whh_b, whhFB, 8, HH);
    }

    // 1) gather embeddings to bf16
    embed_kernel<<<(BB * TT + 255) / 256, 256, 0, stream>>>(x, emb, xe);

    // 2) zero emission accumulator
    const int emN = BB * TT * LL;
    zero_kernel<<<(emN + 255) / 256, 256, 0, stream>>>(em, emN);

    // 3) persistent BiGRU (16 WGs: 2 dirs x 8 batch tiles) with fused emissions
    gru_kernel<<<16, 512, 0, stream>>>(xe, wihFF, whhFF, wihFB, whhFB,
                                       bih_f, bhh_f, bih_b, bhh_b, fcw, em);

    // 4) CRF negative log-likelihood -> single scalar
    crf_kernel<<<1, 128, 0, stream>>>(em, tags, fcb, strt, endt, trans,
                                      (float*)d_out);
}
